// ChannelAttention_45586782879779
// MI455X (gfx1250) — compile-verified, matrix operand order checked by reading, not run on hardware
//
#include <hip/hip_runtime.h>

// ---------------------------------------------------------------------------
// CDNA5 / gfx1250 channel-attention pipeline.
// f16 WMMA 16x16x32, fp32 accum, async global->LDS staging, double buffering.
// ---------------------------------------------------------------------------

typedef __attribute__((ext_vector_type(16))) _Float16 v16h;
typedef __attribute__((ext_vector_type(8)))  float    v8f;

union Frag16 { v16h v; unsigned u[8]; };

__device__ __forceinline__ v8f v8f_zero() {
    v8f z = {0.f, 0.f, 0.f, 0.f, 0.f, 0.f, 0.f, 0.f};
    return z;
}

// Low 32 bits of a generic pointer into LDS are the LDS byte offset (ISA 10.2).
__device__ __forceinline__ unsigned lds_off(const void* p) {
    return (unsigned)(uintptr_t)p;
}

// Async copy 16B global -> LDS (GLOBAL_LOAD_ASYNC_TO_LDS_B128, ASYNCcnt).
__device__ __forceinline__ void async_b128(unsigned lds_addr, const void* gptr) {
    unsigned long long ga = (unsigned long long)(uintptr_t)gptr;
    asm volatile("global_load_async_to_lds_b128 %0, %1, off"
                 :: "v"(lds_addr), "v"(ga) : "memory");
}

__device__ __forceinline__ void async_wait0() {
    asm volatile("s_wait_asynccnt 0x0" ::: "memory");
}
__device__ __forceinline__ void async_wait4() {
    asm volatile("s_wait_asynccnt 0x4" ::: "memory");
}

// Load one 16x32 f16 A/B fragment from an LDS tile row-slab.
// p points at &tile[row * ld + kb] with kb = (lane>>4)*8.
// VGPR v holds K pair: k = kb + (v>=4 ? 16 : 0) + (v&3)*2  (ISA 7.12.2).
__device__ __forceinline__ v16h lds_frag16(const _Float16* p) {
    Frag16 f;
#pragma unroll
    for (int r = 0; r < 8; ++r) {
        const int off = ((r & 3) * 2) + ((r >> 2) * 16);
        f.u[r] = *(const unsigned*)(p + off);
    }
    return f.v;
}

__device__ __forceinline__ v8f wmma_f16(v16h a, v16h b, v8f c) {
    return __builtin_amdgcn_wmma_f32_16x16x32_f16(false, a, false, b,
                                                  (short)0, c, false, false);
}

// ---------------------------------------------------------------------------
// Elementwise fp32 -> f16 convert
// ---------------------------------------------------------------------------
__global__ __launch_bounds__(256) void f32_to_f16_kernel(
    const float* __restrict__ src, _Float16* __restrict__ dst, int n)
{
    int i = (blockIdx.x * blockDim.x + threadIdx.x) * 4;
    if (i + 3 < n) {
        float4 v = *(const float4*)(src + i);
        dst[i + 0] = (_Float16)v.x;
        dst[i + 1] = (_Float16)v.y;
        dst[i + 2] = (_Float16)v.z;
        dst[i + 3] = (_Float16)v.w;
    } else {
        for (; i < n; ++i) dst[i] = (_Float16)src[i];
    }
}

__global__ __launch_bounds__(256) void zero_f32_kernel(float* __restrict__ p, int n)
{
    int i = blockIdx.x * blockDim.x + threadIdx.x;
    if (i < n) p[i] = 0.f;
}

// ---------------------------------------------------------------------------
// Generic WMMA GEMM: C[M,N] = A[M,K] * Bt[N,K]^T   (A,Bt f16 row-major)
// Block tile 128x128, BK=32, 256 threads = 8 waves, wave tile 32x64.
// Double-buffered async global->LDS staging (4 b128/thread in flight).
// ---------------------------------------------------------------------------
template<bool OUT_F16>
__global__ __launch_bounds__(256) void wmma_gemm_kernel(
    const _Float16* __restrict__ A,
    const _Float16* __restrict__ Bt,
    _Float16* __restrict__ Ch,
    float* __restrict__ Cf,
    const float* __restrict__ bias,
    int M, int N, int K)
{
    constexpr int BK  = 32;
    constexpr int LDT = 40;             // 32 + 8 pad halves (80B rows, 16B aligned)
    __shared__ _Float16 As[2][128 * LDT];
    __shared__ _Float16 Bs[2][128 * LDT];

    const int tid  = threadIdx.x;
    const int lane = tid & 31;
    const int wave = tid >> 5;

    const int block_m = blockIdx.y * 128;
    const int block_n = blockIdx.x * 128;

    const int wm = (wave & 3) * 32;     // 4 waves along M
    const int wn = (wave >> 2) * 64;    // 2 waves along N

    const int mf = lane & 15;
    const int kb = (lane >> 4) * 8;
    const int mb = (lane >> 4) * 8;

    // staging coords: chunks tid and tid+256 of 512 16B chunks (128 rows x 4)
    const int r0 = tid >> 2;
    const int c0 = (tid & 3) * 8;
    const int r1 = (tid + 256) >> 2;
    const int c1 = ((tid + 256) & 3) * 8;

    v8f acc[2][4];
#pragma unroll
    for (int i = 0; i < 2; ++i)
#pragma unroll
        for (int j = 0; j < 4; ++j) acc[i][j] = v8f_zero();

    auto issue_stage = [&](int buf, int k0) {
        async_b128(lds_off(&As[buf][r0 * LDT + c0]),
                   A + (size_t)(block_m + r0) * K + k0 + c0);
        async_b128(lds_off(&Bs[buf][r0 * LDT + c0]),
                   Bt + (size_t)(block_n + r0) * K + k0 + c0);
        async_b128(lds_off(&As[buf][r1 * LDT + c1]),
                   A + (size_t)(block_m + r1) * K + k0 + c1);
        async_b128(lds_off(&Bs[buf][r1 * LDT + c1]),
                   Bt + (size_t)(block_n + r1) * K + k0 + c1);
    };

    const int niter = K / BK;
    issue_stage(0, 0);
    int buf = 0;

    for (int i = 0; i < niter; ++i) {
        if (i + 1 < niter) {
            issue_stage(buf ^ 1, (i + 1) * BK);   // prefetch next tile
            async_wait4();                        // wait only for current tile
        } else {
            async_wait0();
        }
        __syncthreads();                          // all waves' tile data landed

        v16h af[2];
#pragma unroll
        for (int tm = 0; tm < 2; ++tm)
            af[tm] = lds_frag16(&As[buf][(wm + tm * 16 + mf) * LDT + kb]);
#pragma unroll
        for (int tn = 0; tn < 4; ++tn) {
            v16h bf = lds_frag16(&Bs[buf][(wn + tn * 16 + mf) * LDT + kb]);
#pragma unroll
            for (int tm = 0; tm < 2; ++tm)
                acc[tm][tn] = wmma_f16(af[tm], bf, acc[tm][tn]);
        }
        __syncthreads();                          // reads done before overwrite
        buf ^= 1;
    }

#pragma unroll
    for (int tm = 0; tm < 2; ++tm) {
#pragma unroll
        for (int tn = 0; tn < 4; ++tn) {
            const int n = block_n + wn + tn * 16 + mf;
#pragma unroll
            for (int r = 0; r < 8; ++r) {
                const size_t m = (size_t)(block_m + wm + tm * 16 + mb + r);
                if (OUT_F16) {
                    Ch[m * (size_t)N + n] = (_Float16)acc[tm][tn][r];
                } else {
                    Cf[m * (size_t)N + n] = acc[tm][tn][r] + (bias ? bias[n] : 0.f);
                }
            }
        }
    }
}

// ---------------------------------------------------------------------------
// S[bh,d,e] += SCALE * sum_n K[n,d] * V[n,e]   (split over n, fp32 atomics)
// grid (32, NSPLIT), 256 threads = 8 waves.
// Transposed staging packs token pairs -> ds_store_b32 (half the DS ops).
// ---------------------------------------------------------------------------
__global__ __launch_bounds__(256) void kv_reduce_kernel(
    const _Float16* __restrict__ QKVh,   // [32768, 1536]
    float* __restrict__ Sf,              // [32, 64, 64] pre-zeroed
    int nsplit)
{
    constexpr int LDT = 40;
    __shared__ _Float16 Ks[64 * LDT];    // [d][token]
    __shared__ _Float16 Vs[64 * LDT];    // [e][token]

    const int tid  = threadIdx.x;
    const int lane = tid & 31;
    const int wave = tid >> 5;
    const int bh    = blockIdx.x;
    const int bIdx  = bh >> 3;
    const int h     = bh & 7;
    const int split = blockIdx.y;

    const int dt  = wave & 3;
    const int et0 = (wave >> 2) * 2;
    const int mf  = lane & 15;
    const int kb  = (lane >> 4) * 8;

    v8f acc0 = v8f_zero();
    v8f acc1 = v8f_zero();

    const int tok_per_split = 8192 / nsplit;
    const size_t rowbase = (size_t)bIdx * 8192 + (size_t)split * tok_per_split;

    // loader role: threads 0..127 stage K, 128..255 stage V; each thread owns
    // one token-pair (2 tokens) x one 8-half chunk, packed into 8 b32 stores.
    const int sel   = tid >> 7;          // 0 = K, 1 = V
    const int rem   = tid & 127;
    const int tpair = rem >> 3;          // 0..15 -> tokens 2*tpair, 2*tpair+1
    const int chunk = rem & 7;           // dims chunk*8 .. +7
    _Float16* dst = sel ? Vs : Ks;
    const int coloff = sel ? 1024 : 512;

    const int steps = tok_per_split / 32;
    for (int s = 0; s < steps; ++s) {
        const size_t row = rowbase + s * 32 + tpair * 2;
        const _Float16* gp = QKVh + row * 1536 + coloff + h * 64 + chunk * 8;
        uint4 d0 = *(const uint4*)gp;
        uint4 d1 = *(const uint4*)(gp + 1536);
        if (s + 1 < steps) __builtin_prefetch(gp + (size_t)32 * 1536, 0, 0);
        const unsigned short* a0 = (const unsigned short*)&d0;
        const unsigned short* a1 = (const unsigned short*)&d1;
#pragma unroll
        for (int j = 0; j < 8; ++j) {    // [dim][tok,tok+1] packed pair
            *(unsigned*)&dst[(chunk * 8 + j) * LDT + tpair * 2] =
                (unsigned)a0[j] | ((unsigned)a1[j] << 16);
        }
        __syncthreads();

        v16h af = lds_frag16(&Ks[(dt * 16 + mf) * LDT + kb]);
        v16h b0 = lds_frag16(&Vs[(et0 * 16 + mf) * LDT + kb]);
        v16h b1 = lds_frag16(&Vs[((et0 + 1) * 16 + mf) * LDT + kb]);
        acc0 = wmma_f16(af, b0, acc0);
        acc1 = wmma_f16(af, b1, acc1);
        __syncthreads();
    }

    float* Sp = Sf + (size_t)bh * 4096;
    const int mb = (lane >> 4) * 8;
    const float scale = 0.125f;          // HEAD_DIM^-0.5
#pragma unroll
    for (int r = 0; r < 8; ++r) {
        const int d = dt * 16 + mb + r;
        atomicAdd(Sp + d * 64 + et0 * 16 + mf,       acc0[r] * scale);
        atomicAdd(Sp + d * 64 + (et0 + 1) * 16 + mf, acc1[r] * scale);
    }
}

// ---------------------------------------------------------------------------
// Row softmax over e for each [bh, d], write f16 attn. grid 32, block 64.
// ---------------------------------------------------------------------------
__global__ __launch_bounds__(64) void softmax_kernel(
    const float* __restrict__ Sf, _Float16* __restrict__ attn_h)
{
    const int bh = blockIdx.x;
    const int d  = threadIdx.x;
    const float* row = Sf + (size_t)bh * 4096 + d * 64;

    float mx = -3.402823466e+38f;
#pragma unroll 8
    for (int e = 0; e < 64; ++e) mx = fmaxf(mx, row[e]);

    float vals[64];
    float sum = 0.f;
#pragma unroll 8
    for (int e = 0; e < 64; ++e) {
        float v = __expf(row[e] - mx);
        vals[e] = v;
        sum += v;
    }
    const float inv = 1.f / sum;
    _Float16* o = attn_h + (size_t)bh * 4096 + d * 64;
#pragma unroll 8
    for (int e = 0; e < 64; ++e) o[e] = (_Float16)(vals[e] * inv);
}

// ---------------------------------------------------------------------------
// O[n, h*64+d] = sum_e attn[d,e] * Q[n,e]  per (b,h); 128 tokens / block.
// grid (32, 64), 256 threads = 8 waves; async staging of Q and attn tiles.
// ---------------------------------------------------------------------------
__global__ __launch_bounds__(256) void attn_out_kernel(
    const _Float16* __restrict__ QKVh,
    const _Float16* __restrict__ attn_h,
    _Float16* __restrict__ Oh)
{
    constexpr int LQ = 72;               // 64 + 8 pad (144B rows, 16B aligned)
    __shared__ _Float16 Qs[128 * LQ];
    __shared__ _Float16 Ws[64 * LQ];

    const int tid  = threadIdx.x;
    const int lane = tid & 31;
    const int wave = tid >> 5;
    const int bh   = blockIdx.x;
    const int bIdx = bh >> 3;
    const int h    = bh & 7;
    const int mblk = blockIdx.y;

    const size_t row0 = (size_t)bIdx * 8192 + (size_t)mblk * 128;

    // Q tile 128x64: 1024 16B chunks, 4 async per thread
#pragma unroll
    for (int c = 0; c < 4; ++c) {
        const int ch  = tid + c * 256;
        const int r   = ch >> 3;
        const int col = (ch & 7) * 8;
        async_b128(lds_off(&Qs[r * LQ + col]),
                   QKVh + (row0 + r) * 1536 + h * 64 + col);
    }
    // attn tile 64x64 (Bt layout: row=d, k=e contiguous): 512 chunks, 2/thread
#pragma unroll
    for (int c = 0; c < 2; ++c) {
        const int ch  = tid + c * 256;
        const int r   = ch >> 3;
        const int col = (ch & 7) * 8;
        async_b128(lds_off(&Ws[r * LQ + col]),
                   attn_h + (size_t)bh * 4096 + r * 64 + col);
    }
    async_wait0();
    __syncthreads();

    const int mf = lane & 15;
    const int kb = (lane >> 4) * 8;
    v8f acc[4];
#pragma unroll
    for (int t = 0; t < 4; ++t) acc[t] = v8f_zero();

#pragma unroll
    for (int ks = 0; ks < 2; ++ks) {     // K = 64 = two 32-slabs
        const int ko = ks * 32;
        v16h af = lds_frag16(&Qs[(wave * 16 + mf) * LQ + ko + kb]);
#pragma unroll
        for (int tn = 0; tn < 4; ++tn) {
            v16h bf = lds_frag16(&Ws[(tn * 16 + mf) * LQ + ko + kb]);
            acc[tn] = wmma_f16(af, bf, acc[tn]);
        }
    }

    const int mb = (lane >> 4) * 8;
#pragma unroll
    for (int tn = 0; tn < 4; ++tn) {
#pragma unroll
        for (int r = 0; r < 8; ++r) {
            const size_t m = row0 + wave * 16 + mb + r;
            Oh[m * 512 + h * 64 + tn * 16 + mf] = (_Float16)acc[tn][r];
        }
    }
}

// ---------------------------------------------------------------------------
// Host launcher
// ---------------------------------------------------------------------------
extern "C" void kernel_launch(void* const* d_in, const int* in_sizes, int n_in,
                              void* d_out, int out_size, void* d_ws, size_t ws_size,
                              hipStream_t stream)
{
    const float* x     = (const float*)d_in[0];   // [4,8,32,32,512]
    const float* Wqkv  = (const float*)d_in[1];   // [1536,512]
    const float* Wproj = (const float*)d_in[2];   // [512,512]
    const float* bproj = (const float*)d_in[3];   // [512]
    float* out = (float*)d_out;                   // fp32, 16,777,216

    const int M  = 32768;                         // B * D*H*W tokens
    const int C  = 512;
    const int C3 = 1536;
    const int NSPLIT = 8;

    char* ws = (char*)d_ws;
    size_t off = 0;
    auto carve = [&](size_t bytes) -> char* {
        char* p = ws + off;
        off = (off + bytes + 255) & ~(size_t)255;
        return p;
    };
    _Float16* Xh     = (_Float16*)carve((size_t)M * C  * 2);   // 32 MB
    _Float16* Wqkvh  = (_Float16*)carve((size_t)C3 * C * 2);   // 1.5 MB
    _Float16* Wprojh = (_Float16*)carve((size_t)C * C  * 2);   // 0.5 MB
    _Float16* QKVh   = (_Float16*)carve((size_t)M * C3 * 2);   // 96 MB
    float*    Sf     = (float*)   carve((size_t)32 * 4096 * 4);
    _Float16* Attnh  = (_Float16*)carve((size_t)32 * 4096 * 2);
    _Float16* Oh     = (_Float16*)carve((size_t)M * C  * 2);   // 32 MB

    // 1) fp32 -> f16 conversions
    {
        int n = M * C;
        f32_to_f16_kernel<<<(n / 4 + 255) / 256, 256, 0, stream>>>(x, Xh, n);
        n = C3 * C;
        f32_to_f16_kernel<<<(n / 4 + 255) / 256, 256, 0, stream>>>(Wqkv, Wqkvh, n);
        n = C * C;
        f32_to_f16_kernel<<<(n / 4 + 255) / 256, 256, 0, stream>>>(Wproj, Wprojh, n);
    }

    // 2) QKV = Xh @ Wqkvh^T  (f16 out)
    {
        dim3 grid(C3 / 128, M / 128);
        wmma_gemm_kernel<true><<<grid, 256, 0, stream>>>(
            Xh, Wqkvh, QKVh, (float*)nullptr, (const float*)nullptr, M, C3, C);
    }

    // 3) S = scale * K^T V  (split-n atomics into fp32)
    {
        int n = 32 * 4096;
        zero_f32_kernel<<<(n + 255) / 256, 256, 0, stream>>>(Sf, n);
        dim3 grid(32, NSPLIT);
        kv_reduce_kernel<<<grid, 256, 0, stream>>>(QKVh, Sf, NSPLIT);
    }

    // 4) softmax rows -> f16 attn
    softmax_kernel<<<32, 64, 0, stream>>>(Sf, Attnh);

    // 5) O = Q @ attn^T
    {
        dim3 grid(32, 8192 / 128);
        attn_out_kernel<<<grid, 256, 0, stream>>>(QKVh, Attnh, Oh);
    }

    // 6) out = Oh @ Wprojh^T + bproj  (fp32 out)
    {
        dim3 grid(C / 128, M / 128);
        wmma_gemm_kernel<false><<<grid, 256, 0, stream>>>(
            Oh, Wprojh, (_Float16*)nullptr, out, bproj, M, C, C);
    }

    (void)in_sizes; (void)n_in; (void)out_size; (void)ws_size;
}
